// AdaptiveMolecularRegressor_84301618086271
// MI455X (gfx1250) — matrix operand order repaired and checked by reading, not compiled, not run on hardware
//
#include <hip/hip_runtime.h>
#include <hip/hip_bf16.h>

typedef __attribute__((ext_vector_type(16))) _Float16 v16h;
typedef __attribute__((ext_vector_type(8)))  float    v8f;

#define IN_DIM 12
#define HID 48
#define ADME 20
#define EPS 1e-5f

static constexpr int NN = 100000;
static constexpr int NE = 3200000;
static constexpr int NG = 2000;
static constexpr int NTILES = NN / 16;   // 6250, exact

// ---- workspace layout (in floats) ----
static constexpr long long OFF_CNT   = 0;          // 100000
static constexpr long long OFF_MSUM0 = 100000;     // 1,200,000
static constexpr long long OFF_MSUM1 = 1300000;    // 4,800,000
static constexpr long long OFF_PSUM  = 6100000;    // 96,000
static constexpr long long OFF_PMAX  = 6196000;    // 96,000 (uint bits)
static constexpr long long OFF_PN    = 6292000;    // 2,000
static constexpr long long OFF_SUM0  = 6294000;    // 48 each below
static constexpr long long OFF_SQ0   = 6294048;
static constexpr long long OFF_SC0   = 6294096;
static constexpr long long OFF_SH0   = 6294144;
static constexpr long long OFF_SUM1  = 6294192;
static constexpr long long OFF_SQ1   = 6294240;
static constexpr long long OFF_SC1   = 6294288;
static constexpr long long OFF_SH1   = 6294336;
static constexpr long long OFF_HSUM  = 6294384;    // 64 each below
static constexpr long long OFF_HSQ   = 6294448;
static constexpr long long OFF_HSC   = 6294512;
static constexpr long long OFF_HSH   = 6294576;
static constexpr long long ZERO_N    = 6294640;    // zero everything above
static constexpr long long OFF_B0    = 6294656;    // 768 floats  = 1536 halfs (32B aligned)
static constexpr long long OFF_B1    = 6295424;    // 2304 floats = 4608 halfs
static constexpr long long OFF_H0    = 6297728;    // 4,800,000
static constexpr long long OFF_H1    = 11097728;   // 4,800,000
static constexpr long long OFF_Z1    = 15897728;   // 128,000  (total ~64.1 MB)

// -------------------- utility --------------------
__global__ void k_zero(float* __restrict__ p, long long n) {
  long long i = (long long)blockIdx.x * blockDim.x + threadIdx.x;
  long long stride = (long long)gridDim.x * blockDim.x;
  for (; i < n; i += stride) p[i] = 0.0f;
}

// -------------------- weight packing into WMMA B layout --------------------
// B (KxN, f16), this code's k order: lanes 0-15 hold K=0..15 (2 per VGPR), lanes 16-31 hold K=16..31.
// Packed as [tileLinear][lane][16 halfs].
__global__ void k_prepB0(const float* __restrict__ wl, const float* __restrict__ wr,
                         _Float16* __restrict__ B) {
  int tid = blockIdx.x * blockDim.x + threadIdx.x;
  if (tid >= 3 * 32 * 16) return;
  int nt   = tid / 512;
  int r    = tid % 512;
  int lane = r / 16;
  int q    = r % 16;
  int k = ((lane < 16) ? 0 : 16) + q;
  int n = nt * 16 + (lane & 15);
  float w = 0.0f;
  if (k < IN_DIM)       w = wl[n * IN_DIM + k];
  else if (k < 2*IN_DIM) w = wr[n * IN_DIM + (k - IN_DIM)];
  B[tid] = (_Float16)w;
}

__global__ void k_prepB1(const float* __restrict__ wl, const float* __restrict__ wr,
                         _Float16* __restrict__ B) {
  int tid = blockIdx.x * blockDim.x + threadIdx.x;
  if (tid >= 9 * 32 * 16) return;
  int kc   = tid / (3 * 512);
  int rest = tid % (3 * 512);
  int nt   = rest / 512;
  int r    = rest % 512;
  int lane = r / 16;
  int q    = r % 16;
  int k = kc * 32 + ((lane < 16) ? 0 : 16) + q;
  int n = nt * 16 + (lane & 15);
  float w = (k < HID) ? wl[n * HID + k] : wr[n * HID + (k - HID)];
  B[((long long)(kc * 3 + nt) * 32 + lane) * 16 + q] = (_Float16)w;
}

// -------------------- edge scatter (segment sums) --------------------
__global__ void k_scatter0(const float* __restrict__ x, const long long* __restrict__ ei,
                           float* __restrict__ cnt, float* __restrict__ msum) {
  int e = blockIdx.x * blockDim.x + threadIdx.x;
  if (e >= NE) return;
  int s = (int)ei[e];
  int d = (int)ei[NE + e];
  atomicAdd(&cnt[d], 1.0f);
  const float* xr = x + (long long)s * IN_DIM;
  float* mr = msum + (long long)d * IN_DIM;
#pragma unroll
  for (int c = 0; c < IN_DIM; ++c) atomicAdd(&mr[c], xr[c]);
}

__global__ void k_scatter1(const float* __restrict__ h, const long long* __restrict__ ei,
                           float* __restrict__ msum) {
  int tid = blockIdx.x * blockDim.x + threadIdx.x;
  if (tid >= NE * 4) return;
  int e = tid >> 2;
  int part = tid & 3;
  int s = (int)ei[e];
  int d = (int)ei[NE + e];
  const float* hr = h + (long long)s * HID + part * 12;
  float* mr = msum + (long long)d * HID + part * 12;
#pragma unroll
  for (int c = 0; c < 12; ++c) atomicAdd(&mr[c], hr[c]);
}

// -------------------- SAGE linear via WMMA: y = [agg|x] @ [wl;wr]^T + b ------------
// wave-per-tile: 16 nodes x 48 outputs. A k order used here (16x32 MxK, matches B packing):
//  lane<16 : M=lane,   VGPR0..3 = K 0..7,   VGPR4..7 = K 8..15
//  lane>=16: M=lane-16,VGPR0..3 = K 16..23, VGPR4..7 = K 24..31
__global__ __launch_bounds__(64)
void k_sage_l0(const float* __restrict__ x, const float* __restrict__ msum,
               const float* __restrict__ cnt, const _Float16* __restrict__ Bp,
               const float* __restrict__ bias, float* __restrict__ y) {
  int lane = threadIdx.x & 31;
  int tile = blockIdx.x * (blockDim.x >> 5) + (threadIdx.x >> 5);
  int m = lane & 15;
  int node = tile * 16 + m;
  float rinv = 1.0f / fmaxf(cnt[node], 1.0f);
  const float* mrow = msum + (long long)node * IN_DIM;
  const float* xrow = x + (long long)node * IN_DIM;
  bool hi = lane >= 16;
  int klo = hi ? 16 : 0;
  int khi = hi ? 24 : 8;
  v16h a;
#pragma unroll
  for (int i = 0; i < 8; ++i) {
    int k0 = klo + i, k1 = khi + i;
    float f0 = (k0 < IN_DIM) ? mrow[k0] * rinv : ((k0 < 2*IN_DIM) ? xrow[k0 - IN_DIM] : 0.0f);
    float f1 = (k1 < IN_DIM) ? mrow[k1] * rinv : ((k1 < 2*IN_DIM) ? xrow[k1 - IN_DIM] : 0.0f);
    a[i]     = (_Float16)f0;
    a[8 + i] = (_Float16)f1;
  }
  const v16h* bv = (const v16h*)Bp;
  v16h b0 = bv[0 * 32 + lane];
  v16h b1 = bv[1 * 32 + lane];
  v16h b2 = bv[2 * 32 + lane];
  v8f c0 = {}, c1 = {}, c2 = {};
  c0 = __builtin_amdgcn_wmma_f32_16x16x32_f16(false, a, false, b0, (short)0, c0, false, false);
  c1 = __builtin_amdgcn_wmma_f32_16x16x32_f16(false, a, false, b1, (short)0, c1, false, false);
  c2 = __builtin_amdgcn_wmma_f32_16x16x32_f16(false, a, false, b2, (short)0, c2, false, false);
  int n0 = lane & 15;
  int rbase = hi ? 8 : 0;
#pragma unroll
  for (int r = 0; r < 8; ++r) {
    float* yr = y + (long long)(tile * 16 + rbase + r) * HID;
    yr[n0]      = c0[r] + bias[n0];
    yr[16 + n0] = c1[r] + bias[16 + n0];
    yr[32 + n0] = c2[r] + bias[32 + n0];
  }
}

__global__ __launch_bounds__(64)
void k_sage_l1(const float* __restrict__ h0, const float* __restrict__ msum,
               const float* __restrict__ cnt, const _Float16* __restrict__ Bp,
               const float* __restrict__ bias, float* __restrict__ y) {
  int lane = threadIdx.x & 31;
  int tile = blockIdx.x * (blockDim.x >> 5) + (threadIdx.x >> 5);
  int m = lane & 15;
  int node = tile * 16 + m;
  float rinv = 1.0f / fmaxf(cnt[node], 1.0f);
  const float* mrow = msum + (long long)node * HID;
  const float* hrow = h0 + (long long)node * HID;
  bool hi = lane >= 16;
  int klo = hi ? 16 : 0;
  int khi = hi ? 24 : 8;
  const v16h* bv = (const v16h*)Bp;
  v8f c0 = {}, c1 = {}, c2 = {};
#pragma unroll
  for (int kc = 0; kc < 3; ++kc) {
    int kb = kc * 32;
    v16h a;
#pragma unroll
    for (int i = 0; i < 8; ++i) {
      int k0 = kb + klo + i, k1 = kb + khi + i;
      float f0 = (k0 < HID) ? mrow[k0] * rinv : hrow[k0 - HID];
      float f1 = (k1 < HID) ? mrow[k1] * rinv : hrow[k1 - HID];
      a[i]     = (_Float16)f0;
      a[8 + i] = (_Float16)f1;
    }
    v16h b0 = bv[(kc * 3 + 0) * 32 + lane];
    v16h b1 = bv[(kc * 3 + 1) * 32 + lane];
    v16h b2 = bv[(kc * 3 + 2) * 32 + lane];
    c0 = __builtin_amdgcn_wmma_f32_16x16x32_f16(false, a, false, b0, (short)0, c0, false, false);
    c1 = __builtin_amdgcn_wmma_f32_16x16x32_f16(false, a, false, b1, (short)0, c1, false, false);
    c2 = __builtin_amdgcn_wmma_f32_16x16x32_f16(false, a, false, b2, (short)0, c2, false, false);
  }
  int n0 = lane & 15;
  int rbase = hi ? 8 : 0;
#pragma unroll
  for (int r = 0; r < 8; ++r) {
    float* yr = y + (long long)(tile * 16 + rbase + r) * HID;
    yr[n0]      = c0[r] + bias[n0];
    yr[16 + n0] = c1[r] + bias[16 + n0];
    yr[32 + n0] = c2[r] + bias[32 + n0];
  }
}

// -------------------- BatchNorm --------------------
__global__ void k_bn_stats48(const float* __restrict__ y, float* __restrict__ gsum,
                             float* __restrict__ gsq, long long n) {
  __shared__ float ss[48];
  __shared__ float sq[48];
  int t = threadIdx.x;
  if (t < 48) { ss[t] = 0.0f; sq[t] = 0.0f; }
  __syncthreads();
  long long i = (long long)blockIdx.x * blockDim.x + t;
  long long stride = (long long)gridDim.x * blockDim.x;
  for (; i < n; i += stride) {
    float v = y[i];
    int c = (int)(i % 48);
    atomicAdd(&ss[c], v);
    atomicAdd(&sq[c], v * v);
  }
  __syncthreads();
  if (t < 48) { atomicAdd(&gsum[t], ss[t]); atomicAdd(&gsq[t], sq[t]); }
}

__global__ void k_bn_finalize(const float* __restrict__ sum, const float* __restrict__ sq,
                              const float* __restrict__ gam, const float* __restrict__ bet,
                              float* __restrict__ scale, float* __restrict__ shift,
                              float inv_count, int nchan) {
  int t = blockIdx.x * blockDim.x + threadIdx.x;
  if (t >= nchan) return;
  float m = sum[t] * inv_count;
  float v = sq[t] * inv_count - m * m;
  float s = gam[t] * rsqrtf(v + EPS);
  scale[t] = s;
  shift[t] = bet[t] - m * s;
}

__global__ void k_bn_apply_relu(float* __restrict__ y, const float* __restrict__ scale,
                                const float* __restrict__ shift, long long n) {
  long long i = (long long)blockIdx.x * blockDim.x + threadIdx.x;
  if (i >= n) return;
  int c = (int)(i % 48);
  y[i] = fmaxf(y[i] * scale[c] + shift[c], 0.0f);
}

// layer-1 apply fused with graph pooling (mean-sum, bitwise max on non-neg floats, counts)
__global__ void k_apply_pool(float* __restrict__ y, const float* __restrict__ scale,
                             const float* __restrict__ shift, const long long* __restrict__ batch,
                             float* __restrict__ psum, unsigned int* __restrict__ pmax,
                             float* __restrict__ pn, long long n) {
  long long i = (long long)blockIdx.x * blockDim.x + threadIdx.x;
  if (i >= n) return;
  int c = (int)(i % 48);
  long long node = i / 48;
  float h = fmaxf(y[i] * scale[c] + shift[c], 0.0f);
  y[i] = h;
  int g = (int)batch[node];
  atomicAdd(&psum[(long long)g * 48 + c], h);
  atomicMax(&pmax[(long long)g * 48 + c], __float_as_uint(h));
  if (c == 0) atomicAdd(&pn[g], 1.0f);
}

// -------------------- head --------------------
__global__ void k_head1(const float* __restrict__ psum, const unsigned int* __restrict__ pmax,
                        const float* __restrict__ pn, const float* __restrict__ adme,
                        const float* __restrict__ hw1, const float* __restrict__ hb1,
                        float* __restrict__ z1, float* __restrict__ hsum, float* __restrict__ hsq) {
  int g = blockIdx.x * blockDim.x + threadIdx.x;
  if (g >= NG) return;
  float comb[2 * HID + ADME];
  float rn = 1.0f / fmaxf(pn[g], 1.0f);
  for (int c = 0; c < 48; ++c) {
    comb[c]      = psum[(long long)g * 48 + c] * rn;
    comb[48 + c] = __uint_as_float(pmax[(long long)g * 48 + c]);
  }
  for (int c = 0; c < ADME; ++c) comb[96 + c] = adme[g * ADME + c];
  for (int o = 0; o < 64; ++o) {
    float acc = hb1[o];
    const float* w = hw1 + o * (2 * HID + ADME);
    for (int k = 0; k < 2 * HID + ADME; ++k) acc += comb[k] * w[k];
    z1[(long long)g * 64 + o] = acc;
    atomicAdd(&hsum[o], acc);
    atomicAdd(&hsq[o], acc * acc);
  }
}

__global__ void k_head2(const float* __restrict__ z1, const float* __restrict__ hscale,
                        const float* __restrict__ hshift, const float* __restrict__ hw2,
                        const float* __restrict__ hb2, const float* __restrict__ hw3,
                        const float* __restrict__ hb3, float* __restrict__ out) {
  int g = blockIdx.x * blockDim.x + threadIdx.x;
  if (g >= NG) return;
  float a[64];
  for (int o = 0; o < 64; ++o)
    a[o] = fmaxf(z1[(long long)g * 64 + o] * hscale[o] + hshift[o], 0.0f);
  float res = hb3[0];
  for (int j = 0; j < 32; ++j) {
    float acc = hb2[j];
    const float* w = hw2 + j * 64;
    for (int o = 0; o < 64; ++o) acc += a[o] * w[o];
    res += fmaxf(acc, 0.0f) * hw3[j];
  }
  out[g] = res;
}

// -------------------- launch --------------------
extern "C" void kernel_launch(void* const* d_in, const int* in_sizes, int n_in,
                              void* d_out, int out_size, void* d_ws, size_t ws_size,
                              hipStream_t stream) {
  const float*     x     = (const float*)d_in[0];
  const long long* ei    = (const long long*)d_in[1];
  const long long* batch = (const long long*)d_in[2];
  const float*     adme  = (const float*)d_in[3];
  const float *wl0=(const float*)d_in[4], *wr0=(const float*)d_in[5], *bc0=(const float*)d_in[6];
  const float *g0 =(const float*)d_in[7], *be0=(const float*)d_in[8];
  const float *wl1=(const float*)d_in[9], *wr1=(const float*)d_in[10], *bc1=(const float*)d_in[11];
  const float *g1 =(const float*)d_in[12], *be1=(const float*)d_in[13];
  const float *hw1=(const float*)d_in[14], *hb1=(const float*)d_in[15];
  const float *hg1=(const float*)d_in[16], *hbe1=(const float*)d_in[17];
  const float *hw2=(const float*)d_in[18], *hb2=(const float*)d_in[19];
  const float *hw3=(const float*)d_in[20], *hb3=(const float*)d_in[21];

  float* ws = (float*)d_ws;
  float* cnt   = ws + OFF_CNT;
  float* msum0 = ws + OFF_MSUM0;
  float* msum1 = ws + OFF_MSUM1;
  float* psum  = ws + OFF_PSUM;
  unsigned int* pmax = (unsigned int*)(ws + OFF_PMAX);
  float* pn    = ws + OFF_PN;
  float* h0    = ws + OFF_H0;
  float* h1    = ws + OFF_H1;
  float* z1    = ws + OFF_Z1;
  _Float16* B0p = (_Float16*)(ws + OFF_B0);
  _Float16* B1p = (_Float16*)(ws + OFF_B1);

  // zero accumulators/stats/pools
  k_zero<<<2048, 256, 0, stream>>>(ws, ZERO_N);

  // pack weights into WMMA B-fragment layout
  k_prepB0<<<6, 256, 0, stream>>>(wl0, wr0, B0p);
  k_prepB1<<<18, 256, 0, stream>>>(wl1, wr1, B1p);

  // layer 0: scatter -> fused [agg|x]@[wl;wr]^T (WMMA) -> BN stats -> apply+ReLU
  k_scatter0<<<(NE + 255) / 256, 256, 0, stream>>>(x, ei, cnt, msum0);
  k_sage_l0<<<NTILES / 2, 64, 0, stream>>>(x, msum0, cnt, B0p, bc0, h0);
  k_bn_stats48<<<1024, 256, 0, stream>>>(h0, ws + OFF_SUM0, ws + OFF_SQ0, (long long)NN * HID);
  k_bn_finalize<<<1, 64, 0, stream>>>(ws + OFF_SUM0, ws + OFF_SQ0, g0, be0,
                                      ws + OFF_SC0, ws + OFF_SH0, 1.0f / NN, 48);
  k_bn_apply_relu<<<((long long)NN * HID + 255) / 256, 256, 0, stream>>>(
      h0, ws + OFF_SC0, ws + OFF_SH0, (long long)NN * HID);

  // layer 1
  k_scatter1<<<(NE * 4 + 255) / 256, 256, 0, stream>>>(h0, ei, msum1);
  k_sage_l1<<<NTILES / 2, 64, 0, stream>>>(h0, msum1, cnt, B1p, bc1, h1);
  k_bn_stats48<<<1024, 256, 0, stream>>>(h1, ws + OFF_SUM1, ws + OFF_SQ1, (long long)NN * HID);
  k_bn_finalize<<<1, 64, 0, stream>>>(ws + OFF_SUM1, ws + OFF_SQ1, g1, be1,
                                      ws + OFF_SC1, ws + OFF_SH1, 1.0f / NN, 48);
  k_apply_pool<<<((long long)NN * HID + 255) / 256, 256, 0, stream>>>(
      h1, ws + OFF_SC1, ws + OFF_SH1, batch, psum, pmax, pn, (long long)NN * HID);

  // head
  k_head1<<<(NG + 63) / 64, 64, 0, stream>>>(psum, pmax, pn, adme, hw1, hb1,
                                             z1, ws + OFF_HSUM, ws + OFF_HSQ);
  k_bn_finalize<<<1, 64, 0, stream>>>(ws + OFF_HSUM, ws + OFF_HSQ, hg1, hbe1,
                                      ws + OFF_HSC, ws + OFF_HSH, 1.0f / NG, 64);
  k_head2<<<(NG + 63) / 64, 64, 0, stream>>>(z1, ws + OFF_HSC, ws + OFF_HSH,
                                             hw2, hb2, hw3, hb3, (float*)d_out);
}
